// MarketHead_66417374265587
// MI455X (gfx1250) — compile-verified, matrix operand order checked by reading, not run on hardware
//
#include <hip/hip_runtime.h>

// ---------------------------------------------------------------------------
// MarketHead fused pipeline for gfx1250 (CDNA5, wave32, WMMA).
// Main GEMM [16384,1031]x[1031,1031] done with split-precision bf16
// (x = hi + lo, acc += hi*hi + hi*lo + lo*hi) via v_wmma_f32_16x16x32_bf16,
// giving ~f32 accuracy at ~2.7x the f32-WMMA (16x16x4) issue efficiency.
// B tiles (pre-split bf16 weights) are staged with CDNA5 async global->LDS
// copies (ASYNCcnt); A tiles are converted f32->bf16 hi/lo in VALU on the
// way into LDS.
// Second GEMM (N=5) fused into the epilogue with half-wave reductions +
// f32 global atomics, so no hidden-activation buffer is materialized.
// ---------------------------------------------------------------------------

#define DEV __device__ __forceinline__

constexpr int Dv   = 1024;
constexpr int Fv   = 1031;            // D + 7 feature columns
constexpr int KPAD = 1056;            // 33 * 32
constexpr int NPAD = 1152;            // 9 * 128
constexpr int BM = 128, BN = 128, BK = 32;
constexpr int KSTEPS = KPAD / BK;     // 33
constexpr int LDT = 40;               // LDS row pitch in bf16 elems: 80B, 16B-aligned, conflict-free

typedef __attribute__((ext_vector_type(16))) __bf16 v16bf;
typedef __attribute__((ext_vector_type(8)))  __bf16 v8bf;
typedef __attribute__((ext_vector_type(8)))  float  v8f;
typedef int v4i __attribute__((vector_size(16)));   // matches builtin param type

#if defined(__HIP_DEVICE_COMPILE__) && __has_builtin(__builtin_amdgcn_global_load_async_to_lds_b128)
#define HAVE_ASYNC_LDS 1
#else
#define HAVE_ASYNC_LDS 0
#endif

DEV unsigned short f32_to_bf16(float x) {
  unsigned int u = __float_as_uint(x);
  u += 0x7fffu + ((u >> 16) & 1u);    // round-to-nearest-even
  return (unsigned short)(u >> 16);
}
DEV float bf16_to_f32(unsigned short h) {
  return __uint_as_float(((unsigned int)h) << 16);
}

// 16B global -> LDS copy: async DMA path on CDNA5, VGPR bounce otherwise.
DEV void copy16_g2l(const unsigned short* __restrict__ g,
                    unsigned short* __restrict__ l) {
#if HAVE_ASYNC_LDS
  __builtin_amdgcn_global_load_async_to_lds_b128(
      (__attribute__((address_space(1))) v4i*)(g),
      (__attribute__((address_space(3))) v4i*)(l),
      0 /*offset*/, 0 /*cpol*/);
#else
  *(uint4*)l = *(const uint4*)g;
#endif
}

DEV void wait_async_lds() {
#if HAVE_ASYNC_LDS
#if __has_builtin(__builtin_amdgcn_s_wait_asynccnt)
  __builtin_amdgcn_s_wait_asynccnt(0);
#else
  asm volatile("s_wait_asynccnt 0x0" ::: "memory");
#endif
#endif
}

// Load one v16bf fragment: two contiguous 16B chunks (K-packing per ISA layout).
DEV v16bf load_frag(const unsigned short* p) {
  v8bf a = *(const v8bf*)p;
  v8bf b = *(const v8bf*)(p + 16);
  return __builtin_shufflevector(a, b, 0,1,2,3,4,5,6,7,8,9,10,11,12,13,14,15);
}

// ---------------------------------------------------------------------------
// Kernel A: W1 [F,F] f32 row-major -> transposed, padded bf16 hi/lo planes
// w1t_{hi,lo}[n*KPAD + k] = split(W1[k*F + n]),  zero outside [F,F].
// ---------------------------------------------------------------------------
__global__ void split_w1_kernel(const float* __restrict__ w1,
                                unsigned short* __restrict__ w1t_hi,
                                unsigned short* __restrict__ w1t_lo) {
  size_t idx = (size_t)blockIdx.x * blockDim.x + threadIdx.x;
  if (idx >= (size_t)NPAD * KPAD) return;
  int n = (int)(idx / KPAD);
  int k = (int)(idx % KPAD);
  float v = (n < Fv && k < Fv) ? w1[(size_t)k * Fv + n] : 0.0f;
  unsigned short h = f32_to_bf16(v);
  w1t_hi[idx] = h;
  w1t_lo[idx] = f32_to_bf16(v - bf16_to_f32(h));
}

// ---------------------------------------------------------------------------
// Kernel B: per-row reductions -> 8-float scalar record
// scal[r*8 + {0..7}] = {align, div, tens, dist, r_current, r_basin, r_final, 0}
// ---------------------------------------------------------------------------
__global__ void row_scalars_kernel(const float* __restrict__ hf,
                                   const float* __restrict__ vc,
                                   const float* __restrict__ vb,
                                   float* __restrict__ scal) {
  const int r   = blockIdx.x;
  const int tid = threadIdx.x;               // 256 threads, 1 float4 each
  const float4 h = ((const float4*)(hf + (size_t)r * Dv))[tid];
  const float4 c = ((const float4*)(vc + (size_t)r * Dv))[tid];
  const float4 b = ((const float4*)(vb + (size_t)r * Dv))[tid];

  float s[5];
  s[0] = h.x*h.x + h.y*h.y + h.z*h.z + h.w*h.w;                 // sum h^2
  s[1] = c.x*c.x + c.y*c.y + c.z*c.z + c.w*c.w;                 // sum c^2
  s[2] = b.x*b.x + b.y*b.y + b.z*b.z + b.w*b.w;                 // sum b^2
  s[3] = c.x*b.x + c.y*b.y + c.z*b.z + c.w*b.w;                 // sum c*b
  float dx = c.x-b.x, dy = c.y-b.y, dz = c.z-b.z, dw = c.w-b.w;
  s[4] = dx*dx + dy*dy + dz*dz + dw*dw;                         // sum (c-b)^2

  #pragma unroll
  for (int i = 0; i < 5; ++i)
    #pragma unroll
    for (int off = 1; off < 32; off <<= 1)
      s[i] += __shfl_xor(s[i], off, 32);

  __shared__ float red[8][5];
  const int wave = tid >> 5, lane = tid & 31;
  if (lane == 0)
    for (int i = 0; i < 5; ++i) red[wave][i] = s[i];
  __syncthreads();

  if (tid == 0) {
    float t[5] = {0.f, 0.f, 0.f, 0.f, 0.f};
    for (int w = 0; w < 8; ++w)
      for (int i = 0; i < 5; ++i) t[i] += red[w][i];
    const float EPS = 1e-12f;
    float rf = sqrtf(t[0]);
    float rc = sqrtf(t[1]);
    float rb = sqrtf(t[2]);
    float mc = fmaxf(rc, EPS), mb = fmaxf(rb, EPS);
    float align = t[3] / (mc * mb);
    float nc2 = t[1] / (mc * mc);
    float nb2 = t[2] / (mb * mb);
    float div = sqrtf(fmaxf(nc2 + nb2 - 2.0f * align, 0.0f));
    float* o = scal + (size_t)r * 8;
    o[0] = align;
    o[1] = div;
    o[2] = div * div;        // tension
    o[3] = sqrtf(t[4]);      // dist
    o[4] = rc;
    o[5] = rb;
    o[6] = rf;
    o[7] = 0.0f;             // K padding (col 1031)
  }
}

// ---------------------------------------------------------------------------
// Kernel C: out[b,j] = b2[j]  (atomically accumulated into by the GEMM)
// ---------------------------------------------------------------------------
__global__ void init_out_kernel(float* __restrict__ out,
                                const float* __restrict__ b2, int n) {
  int i = blockIdx.x * blockDim.x + threadIdx.x;
  if (i < n) out[i] = b2[i % 5];
}

// ---------------------------------------------------------------------------
// Kernel D: fused   out += relu(feat @ W1 + b1) @ W2
// Block tile 128x128, 8 waves (2 along M x 4 along N), wave tile 64x32.
// ---------------------------------------------------------------------------
__global__ __launch_bounds__(256)
void gemm_fused_kernel(const float* __restrict__ hfinal,
                       const float* __restrict__ scal,
                       const unsigned short* __restrict__ w1t_hi,
                       const unsigned short* __restrict__ w1t_lo,
                       const float* __restrict__ b1,
                       const float* __restrict__ w2,
                       float* __restrict__ out) {
  __shared__ unsigned short sA_hi[BM * LDT];
  __shared__ unsigned short sA_lo[BM * LDT];
  __shared__ unsigned short sB_hi[BN * LDT];
  __shared__ unsigned short sB_lo[BN * LDT];

  const int tid  = threadIdx.x;
  const int lane = tid & 31;
  const int wave = tid >> 5;
  const int wm = wave & 1;              // 0..1  -> M offset wm*64
  const int wn = wave >> 1;             // 0..3  -> N offset wn*32
  const int m0 = blockIdx.y * BM;
  const int n0 = blockIdx.x * BN;

  // cooperative-load indices: 2 threads per row, 16 elems per thread
  const int lr  = tid >> 1;             // 0..127
  const int lc0 = (tid & 1) * 16;       // 0 or 16
  const int growA = m0 + lr;            // batch row for A loads
  const int gnB   = n0 + lr;            // W1T row (output column) for B loads

  v8f acc[4][2];
  #pragma unroll
  for (int mt = 0; mt < 4; ++mt)
    #pragma unroll
    for (int nt = 0; nt < 2; ++nt)
      #pragma unroll
      for (int v = 0; v < 8; ++v) acc[mt][nt][v] = 0.0f;

  const int half = lane >> 4;           // 0: lanes 0-15, 1: lanes 16-31
  const int l16  = lane & 15;
  const int kb   = half * 8;            // first K chunk offset (elems)

  for (int ks = 0; ks < KSTEPS; ++ks) {
    const int k0 = ks * BK;

    // ---- stage B tile via async global->LDS (pre-split bf16, [n][k]) ----
    {
      const size_t gofs = (size_t)gnB * KPAD + k0 + lc0;
      const int    dofs = lr * LDT + lc0;
      copy16_g2l(w1t_hi + gofs,     &sB_hi[dofs]);
      copy16_g2l(w1t_hi + gofs + 8, &sB_hi[dofs + 8]);
      copy16_g2l(w1t_lo + gofs,     &sB_lo[dofs]);
      copy16_g2l(w1t_lo + gofs + 8, &sB_lo[dofs + 8]);
    }

    // ---- prefetch next A tile into cache hierarchy (global_prefetch) ----
    if (ks + 1 < KSTEPS) {
      const int kn = k0 + BK;
      if (kn + lc0 + 15 < Dv)
        __builtin_prefetch(hfinal + (size_t)growA * Dv + kn + lc0, 0, 3);
    }

    // ---- stage A tile (f32 -> bf16 hi/lo) ----
    if (k0 + lc0 + 15 < Dv) {          // fast path: fully inside h_final
      const float4* src = (const float4*)(hfinal + (size_t)growA * Dv + k0 + lc0);
      #pragma unroll
      for (int i = 0; i < 4; ++i) {
        float4 v = src[i];
        float x[4] = {v.x, v.y, v.z, v.w};
        unsigned short hs[4], ls[4];
        #pragma unroll
        for (int e = 0; e < 4; ++e) {
          hs[e] = f32_to_bf16(x[e]);
          ls[e] = f32_to_bf16(x[e] - bf16_to_f32(hs[e]));
        }
        const unsigned int hi01 = (unsigned int)hs[0] | ((unsigned int)hs[1] << 16);
        const unsigned int hi23 = (unsigned int)hs[2] | ((unsigned int)hs[3] << 16);
        const unsigned int lo01 = (unsigned int)ls[0] | ((unsigned int)ls[1] << 16);
        const unsigned int lo23 = (unsigned int)ls[2] | ((unsigned int)ls[3] << 16);
        const int c = lc0 + i * 4;
        *(unsigned int*)&sA_hi[lr * LDT + c]     = hi01;
        *(unsigned int*)&sA_hi[lr * LDT + c + 2] = hi23;
        *(unsigned int*)&sA_lo[lr * LDT + c]     = lo01;
        *(unsigned int*)&sA_lo[lr * LDT + c + 2] = lo23;
      }
    } else {                            // last k-step: scalar features + padding
      #pragma unroll
      for (int i = 0; i < 16; ++i) {
        const int kg = k0 + lc0 + i;
        float v = 0.0f;
        if (kg < Dv)        v = hfinal[(size_t)growA * Dv + kg];
        else if (kg < Fv)   v = scal[(size_t)growA * 8 + (kg - Dv)];
        unsigned short h = f32_to_bf16(v);
        sA_hi[lr * LDT + lc0 + i] = h;
        sA_lo[lr * LDT + lc0 + i] = f32_to_bf16(v - bf16_to_f32(h));
      }
    }

    wait_async_lds();                  // ASYNCcnt == 0 before consuming B tile
    __syncthreads();

    // ---- fragments + WMMA ----
    v16bf bh[2], bl[2];
    #pragma unroll
    for (int nt = 0; nt < 2; ++nt) {
      const int nrow = wn * 32 + nt * 16 + l16;
      bh[nt] = load_frag(&sB_hi[nrow * LDT + kb]);
      bl[nt] = load_frag(&sB_lo[nrow * LDT + kb]);
    }
    #pragma unroll
    for (int mt = 0; mt < 4; ++mt) {
      const int arow = wm * 64 + mt * 16 + l16;
      v16bf ah = load_frag(&sA_hi[arow * LDT + kb]);
      v16bf al = load_frag(&sA_lo[arow * LDT + kb]);
      #pragma unroll
      for (int nt = 0; nt < 2; ++nt) {
        acc[mt][nt] = __builtin_amdgcn_wmma_f32_16x16x32_bf16(
            false, ah, false, bh[nt], (short)0, acc[mt][nt], false, false);
        acc[mt][nt] = __builtin_amdgcn_wmma_f32_16x16x32_bf16(
            false, ah, false, bl[nt], (short)0, acc[mt][nt], false, false);
        acc[mt][nt] = __builtin_amdgcn_wmma_f32_16x16x32_bf16(
            false, al, false, bh[nt], (short)0, acc[mt][nt], false, false);
      }
    }
    __syncthreads();
  }

  // ---- fused epilogue: bias + relu + (hidden @ W2) with half-wave reduce ----
  // C/D layout: VGPR v, lanes 0-15 -> M = v, lanes 16-31 -> M = 8+v; N = lane%16.
  #pragma unroll
  for (int mt = 0; mt < 4; ++mt) {
    const int rowBase = m0 + wm * 64 + mt * 16 + half * 8;
    #pragma unroll
    for (int v = 0; v < 8; ++v) {
      const int row = rowBase + v;
      float hv[2];
      int   nn[2];
      #pragma unroll
      for (int nt = 0; nt < 2; ++nt) {
        const int n = n0 + wn * 32 + nt * 16 + l16;
        nn[nt] = n;
        float a = (n < Fv) ? (acc[mt][nt][v] + b1[n]) : 0.0f;
        hv[nt] = a > 0.0f ? a : 0.0f;
      }
      #pragma unroll
      for (int j = 0; j < 5; ++j) {
        float p = 0.0f;
        if (nn[0] < Fv) p += hv[0] * w2[(size_t)nn[0] * 5 + j];
        if (nn[1] < Fv) p += hv[1] * w2[(size_t)nn[1] * 5 + j];
        p += __shfl_xor(p, 1, 16);
        p += __shfl_xor(p, 2, 16);
        p += __shfl_xor(p, 4, 16);
        p += __shfl_xor(p, 8, 16);
        if (l16 == 0) atomicAdd(&out[(size_t)row * 5 + j], p);
      }
    }
  }
}

// ---------------------------------------------------------------------------
extern "C" void kernel_launch(void* const* d_in, const int* in_sizes, int n_in,
                              void* d_out, int out_size, void* d_ws, size_t ws_size,
                              hipStream_t stream) {
  const float* hf = (const float*)d_in[0];   // h_final   [B,1024]
  const float* vc = (const float*)d_in[1];   // v_current [B,1024]
  const float* vb = (const float*)d_in[2];   // v_basin   [B,1024]
  const float* W1 = (const float*)d_in[3];   // [F,F]
  const float* b1 = (const float*)d_in[4];   // [F]
  const float* W2 = (const float*)d_in[5];   // [F,5]
  const float* b2 = (const float*)d_in[6];   // [5]
  float* out = (float*)d_out;                // [B,5]

  const int Brows = in_sizes[0] / Dv;        // 16384

  // workspace layout (16B-aligned segments)
  char* ws = (char*)d_ws;
  const size_t scalBytes = (size_t)Brows * 8 * sizeof(float);           // 512 KB
  const size_t w1Bytes   = (size_t)NPAD * KPAD * sizeof(unsigned short); // 2.32 MB
  float*          scal  = (float*)ws;
  unsigned short* w1hi  = (unsigned short*)(ws + scalBytes);
  unsigned short* w1lo  = (unsigned short*)(ws + scalBytes + w1Bytes);
  (void)n_in; (void)ws_size;

  // A: split + transpose + pad W1 into bf16 hi/lo planes
  {
    const size_t total = (size_t)NPAD * KPAD;
    const int blocks = (int)((total + 255) / 256);
    split_w1_kernel<<<blocks, 256, 0, stream>>>(W1, w1hi, w1lo);
  }
  // B: per-row scalar features
  row_scalars_kernel<<<Brows, 256, 0, stream>>>(hf, vc, vb, scal);
  // C: initialize output with b2 bias
  init_out_kernel<<<(out_size + 255) / 256, 256, 0, stream>>>(out, b2, out_size);
  // D: fused WMMA GEMM + relu + second GEMM via atomics
  {
    dim3 grid(NPAD / BN, Brows / BM);      // (9, 128)
    gemm_fused_kernel<<<grid, 256, 0, stream>>>(hf, scal, w1hi, w1lo, b1, W2, out);
  }
}
